// CWRRTCell_27255862460693
// MI455X (gfx1250) — matrix operand order repaired and checked by reading, not compiled, not run on hardware
//
#include <hip/hip_runtime.h>
#include <hip/hip_bf16.h>

// ---------- types ----------
typedef __attribute__((ext_vector_type(16))) __bf16 v16bf;
typedef __attribute__((ext_vector_type(8)))  float  v8f;
typedef int v4i __attribute__((vector_size(16)));

union FragU { v16bf v; float4 q[2]; };

#define AS1 __attribute__((address_space(1)))
#define AS3 __attribute__((address_space(3)))

#if __has_builtin(__builtin_amdgcn_global_load_async_to_lds_b128)
#define HAVE_ASYNC_LDS 1
#endif

// generic->addrspace pointer helpers (via integer casts: LDS offset = low 32 bits
// of the generic aperture address; global generic == as1 numerically)
__device__ __forceinline__ AS1 v4i* gvcast(const void* p) {
  return (AS1 v4i*)(uintptr_t)p;
}
__device__ __forceinline__ AS3 v4i* lvcast(void* p) {
  return (AS3 v4i*)(unsigned int)(uintptr_t)p;
}
__device__ __forceinline__ void wait_async_lds() {
#if __has_builtin(__builtin_amdgcn_s_wait_asynccnt)
  __builtin_amdgcn_s_wait_asynccnt(0);
#else
  asm volatile("s_wait_asynccnt 0" ::: "memory");
#endif
}

// ---------- WMMA fragment loaders (16x16x32 bf16, contiguous k order shared by A and B) ----------
// A: [16 x 32] tile, row-major source (pointer pre-offset to tile row 0).
// lane<16: row=lane, K = k0+0..15 ; lane>=16: row=lane-16, K = k0+16..31
__device__ __forceinline__ v16bf load_a_frag(const __bf16* A, int lda, int k0, int lane) {
  int r  = lane & 15;
  int hi = lane >> 4;
  const __bf16* p = A + (size_t)r * lda + k0 + hi * 16;
  FragU f;
  f.q[0] = *(const float4*)(p);
  f.q[1] = *(const float4*)(p + 8);
  return f.v;
}
// B: [32 x 16] tile supplied as Bt row-major [N,K] (pointer pre-offset to tile col 0).
// lane<16: col=lane, K = k0+0..15 ; lane>=16: col=lane-16, K = k0+16..31  (contiguous in Bt)
__device__ __forceinline__ v16bf load_b_frag(const __bf16* Bt, int ldb, int k0, int lane) {
  int c  = lane & 15;
  int hi = lane >> 4;
  const __bf16* p = Bt + (size_t)c * ldb + k0 + hi * 16;
  FragU f;
  f.q[0] = *(const float4*)(p);
  f.q[1] = *(const float4*)(p + 8);
  return f.v;
}

__device__ __forceinline__ float wave_sum(float v) {
  #pragma unroll
  for (int off = 16; off >= 1; off >>= 1) v += __shfl_xor(v, off, 32);
  return v;
}

// ---------- kernel 1: weight transpose + cvt to bf16 : dst[c][r] = src[r][c] ----------
__global__ __launch_bounds__(256) void wtrans_kernel(const float* __restrict__ src,
                                                     __bf16* __restrict__ dst,
                                                     int rows, int cols) {
  int idx = blockIdx.x * 256 + threadIdx.x;
  if (idx >= rows * cols) return;
  int r = idx / cols, c = idx - r * cols;
  dst[(size_t)c * rows + r] = (__bf16)src[idx];
}

// ---------- kernel 2: x_in, mem shift (into d_out), fused LN1 -> bf16 ----------
__global__ __launch_bounds__(256) void pre_kernel(const float* __restrict__ mem,
                                                  const float* __restrict__ x,
                                                  const float* __restrict__ delta,
                                                  const float* __restrict__ g,
                                                  const float* __restrict__ be,
                                                  float* __restrict__ out_mem,
                                                  __bf16* __restrict__ y1) {
  __shared__ float rbuf[16];
  int b = blockIdx.x, tid = threadIdx.x;
  const float* mb = mem + (size_t)b * 16 * 512;
  float* ob = out_mem + (size_t)b * 16 * 512;
  for (int i = tid; i < 15 * 512; i += 256) ob[i] = mb[i + 512];
  size_t xo = (size_t)b * 512;
  float x0 = x[xo + tid]       + delta[xo + tid];
  float x1 = x[xo + tid + 256] + delta[xo + tid + 256];
  ob[15 * 512 + tid]       = x0;
  ob[15 * 512 + tid + 256] = x1;
  float s1 = wave_sum(x0 + x1);
  float s2 = wave_sum(x0 * x0 + x1 * x1);
  int lane = tid & 31, wave = tid >> 5;
  if (lane == 0) { rbuf[wave] = s1; rbuf[8 + wave] = s2; }
  __syncthreads();
  float t1 = 0.f, t2 = 0.f;
  #pragma unroll
  for (int i = 0; i < 8; ++i) { t1 += rbuf[i]; t2 += rbuf[8 + i]; }
  float mu  = t1 * (1.f / 512.f);
  float var = t2 * (1.f / 512.f) - mu * mu;
  float inv = rsqrtf(var + 1e-6f);
  y1[xo + tid]       = (__bf16)((x0 - mu) * inv * g[tid]       + be[tid]);
  y1[xo + tid + 256] = (__bf16)((x1 - mu) * inv * g[tid + 256] + be[tid + 256]);
}

// ---------- kernel: standalone LN (for LN2) f32 -> bf16 ----------
__global__ __launch_bounds__(256) void ln_kernel(const float* __restrict__ in,
                                                 const float* __restrict__ g,
                                                 const float* __restrict__ be,
                                                 __bf16* __restrict__ out) {
  __shared__ float rbuf[16];
  int b = blockIdx.x, tid = threadIdx.x;
  size_t xo = (size_t)b * 512;
  float x0 = in[xo + tid], x1 = in[xo + tid + 256];
  float s1 = wave_sum(x0 + x1);
  float s2 = wave_sum(x0 * x0 + x1 * x1);
  int lane = tid & 31, wave = tid >> 5;
  if (lane == 0) { rbuf[wave] = s1; rbuf[8 + wave] = s2; }
  __syncthreads();
  float t1 = 0.f, t2 = 0.f;
  #pragma unroll
  for (int i = 0; i < 8; ++i) { t1 += rbuf[i]; t2 += rbuf[8 + i]; }
  float mu  = t1 * (1.f / 512.f);
  float var = t2 * (1.f / 512.f) - mu * mu;
  float inv = rsqrtf(var + 1e-6f);
  out[xo + tid]       = (__bf16)((x0 - mu) * inv * g[tid]       + be[tid]);
  out[xo + tid + 256] = (__bf16)((x1 - mu) * inv * g[tid + 256] + be[tid + 256]);
}

// ---------- generic WMMA GEMM: C[rows,N] = A[rows,K](bf16) * Bt[N,K](bf16)^T + bias ----------
// A panel staged in LDS (async-to-LDS when available); each wave owns 4 N-tiles so one
// A fragment feeds 4 WMMAs.  Block covers 16 rows x 512 cols.
// mode 0: out_bf16 = (acc+bias)*scale                      (Q projection)
// mode 1: out_f32  = acc+bias+res[row*res_stride+n]        (O proj + residual -> x_mid)
// mode 2: out_bf16 = gelu_tanh(acc+bias)                   (MLP up)
// mode 3: x_out = res + acc+bias -> out_f32 ; ssum_out = ssum_in*0.9 + x_out*0.1
__global__ __launch_bounds__(256) void gemm_bf16_kernel(
    const __bf16* __restrict__ A, const __bf16* __restrict__ Bt,
    const float* __restrict__ bias, int K, int N, int mode, float scale,
    const float* __restrict__ res, long res_stride,
    float* __restrict__ outf, __bf16* __restrict__ outb,
    const float* __restrict__ ssum_in, float* __restrict__ ssum_out) {
  __shared__ __align__(16) __bf16 Atile[16 * 512];  // 16 KB chunk of A
  int tid = threadIdx.x;
  int lane = tid & 31;
  int wave = tid >> 5;
  int m0 = blockIdx.x * 16;
  int n0 = blockIdx.y * 512 + wave * 64;  // 4 consecutive 16-wide tiles per wave
  const __bf16* Bb[4];
  #pragma unroll
  for (int t = 0; t < 4; ++t) Bb[t] = Bt + (size_t)(n0 + t * 16) * K;
  v8f acc[4] = {};

  for (int kc = 0; kc < K; kc += 512) {
    const __bf16* Ac = A + (size_t)m0 * K + kc;
    // stage 16 x 512 bf16 panel (1024 x 16B)
#ifdef HAVE_ASYNC_LDS
    for (int t = tid; t < 1024; t += 256) {
      int rr = t >> 6, cc = (t & 63) * 8;
      __builtin_amdgcn_global_load_async_to_lds_b128(
          gvcast(Ac + (size_t)rr * K + cc), lvcast(&Atile[rr * 512 + cc]), 0, 0);
    }
    wait_async_lds();
#else
    for (int t = tid; t < 1024; t += 256) {
      int rr = t >> 6, cc = (t & 63) * 8;
      *(float4*)&Atile[rr * 512 + cc] = *(const float4*)&Ac[(size_t)rr * K + cc];
    }
#endif
    __syncthreads();
    for (int k0 = 0; k0 < 512; k0 += 32) {
      v16bf af = load_a_frag((const __bf16*)Atile, 512, k0, lane);
      #pragma unroll
      for (int t = 0; t < 4; ++t) {
        v16bf bf = load_b_frag(Bb[t], K, kc + k0, lane);
        acc[t] = __builtin_amdgcn_wmma_f32_16x16x32_bf16(false, af, false, bf, (short)0,
                                                         acc[t], false, false);
      }
    }
    __syncthreads();
  }

  int c = lane & 15;
  int mb = m0 + ((lane >> 4) ? 8 : 0);
  #pragma unroll
  for (int t = 0; t < 4; ++t) {
    int n = n0 + t * 16 + c;
    float bv = bias[n];
    #pragma unroll
    for (int r = 0; r < 8; ++r) {
      int row = mb + r;
      float v = acc[t][r] + bv;
      if (mode == 0) {
        outb[(size_t)row * N + n] = (__bf16)(v * scale);
      } else if (mode == 1) {
        outf[(size_t)row * N + n] = v + res[(size_t)row * res_stride + n];
      } else if (mode == 2) {
        float th = tanhf(0.7978845608028654f * (v + 0.044715f * v * v * v));
        outb[(size_t)row * N + n] = (__bf16)(0.5f * v * (1.f + th));
      } else {
        float xo = res[(size_t)row * res_stride + n] + v;
        outf[(size_t)row * N + n] = xo;
        ssum_out[(size_t)row * N + n] = ssum_in[(size_t)row * N + n] * 0.9f + xo * 0.1f;
      }
    }
  }
}

// ---------- fused per-batch attention: K/V WMMA in LDS + softmax + weighted sum ----------
// LDS map (51712 B): [0,16K) A bf16 | [16K,48K) K|V bf16 (aliased by f32 staging) |
//                    [48K,50K) q f32 | [50K, +512) p
__global__ __launch_bounds__(256) void attn_kernel(const float* __restrict__ mem_out,
                                                   const __bf16* __restrict__ wkT,
                                                   const float* __restrict__ bk,
                                                   const __bf16* __restrict__ wvT,
                                                   const float* __restrict__ bv,
                                                   const __bf16* __restrict__ qin,
                                                   __bf16* __restrict__ o_out) {
  __shared__ __align__(16) unsigned char smem[51712];
  __bf16* Alds = (__bf16*)smem;
  __bf16* Klds = (__bf16*)(smem + 16384);
  __bf16* Vlds = (__bf16*)(smem + 32768);
  float*  stage = (float*)(smem + 16384);   // 32 KB, dead before K/V written
  float*  qlds = (float*)(smem + 49152);
  float (*plds)[16] = (float (*)[16])(smem + 51200);

  int b = blockIdx.x;
  int tid = threadIdx.x, lane = tid & 31, wave = tid >> 5;
  const float* Mb = mem_out + (size_t)b * 16 * 512;

  // stage the 16x512 f32 memory tile (2048 x 16B), then convert to bf16
#ifdef HAVE_ASYNC_LDS
  for (int t = tid; t < 2048; t += 256) {
    __builtin_amdgcn_global_load_async_to_lds_b128(
        gvcast(Mb + t * 4), lvcast(stage + t * 4), 0, 0);
  }
  wait_async_lds();
#else
  for (int t = tid; t < 2048; t += 256)
    *(float4*)(stage + t * 4) = *(const float4*)(Mb + t * 4);
#endif
  __syncthreads();
  for (int i = tid; i < 16 * 512; i += 256) Alds[i] = (__bf16)stage[i];
  for (int i = tid; i < 512; i += 256) qlds[i] = (float)qin[(size_t)b * 512 + i];
  __syncthreads();

  // 64 tile jobs: j<32 -> K tiles, j>=32 -> V tiles; one 16x16 WMMA tile each, Kdim=512
  for (int j = wave; j < 64; j += 8) {
    const __bf16* Bt  = (j < 32) ? wkT : wvT;
    const float*  bia = (j < 32) ? bk  : bv;
    int n0 = (j & 31) * 16;
    const __bf16* Bb = Bt + (size_t)n0 * 512;
    v8f acc = {};
    for (int k0 = 0; k0 < 512; k0 += 32) {
      v16bf af = load_a_frag((const __bf16*)Alds, 512, k0, lane);
      v16bf bf = load_b_frag(Bb, 512, k0, lane);
      acc = __builtin_amdgcn_wmma_f32_16x16x32_bf16(false, af, false, bf, (short)0, acc,
                                                    false, false);
    }
    __bf16* dst = (j < 32) ? Klds : Vlds;
    int c = lane & 15;
    int mb = (lane >> 4) ? 8 : 0;
    float bvv = bia[n0 + c];
    #pragma unroll
    for (int r = 0; r < 8; ++r) dst[(mb + r) * 512 + n0 + c] = (__bf16)(acc[r] + bvv);
  }
  __syncthreads();

  // one wave per head
  int h = wave;
  int m = lane & 15;  // both half-waves compute the same 16 scores
  float s = 0.f;
  #pragma unroll 8
  for (int d = 0; d < 64; ++d) s += qlds[h * 64 + d] * (float)Klds[m * 512 + h * 64 + d];
  float mx = s;
  #pragma unroll
  for (int off = 8; off >= 1; off >>= 1) mx = fmaxf(mx, __shfl_xor(mx, off, 32));
  float e = __expf(s - mx);
  float se = e;
  #pragma unroll
  for (int off = 8; off >= 1; off >>= 1) se += __shfl_xor(se, off, 32);
  if (lane < 16) plds[h][lane] = e / se;
  __syncthreads();

  #pragma unroll
  for (int part = 0; part < 2; ++part) {
    int d = lane + part * 32;
    float acc = 0.f;
    #pragma unroll
    for (int mm = 0; mm < 16; ++mm) acc += plds[h][mm] * (float)Vlds[mm * 512 + h * 64 + d];
    o_out[(size_t)b * 512 + h * 64 + d] = (__bf16)acc;
  }
}

// ---------- host side ----------
extern "C" void kernel_launch(void* const* d_in, const int* in_sizes, int n_in,
                              void* d_out, int out_size, void* d_ws, size_t ws_size,
                              hipStream_t stream) {
  const int B = 8192, E = 512, M = 16, F = 2048;
  const float* mem    = (const float*)d_in[0];
  const float* ssum   = (const float*)d_in[1];
  const float* x      = (const float*)d_in[2];
  const float* delta  = (const float*)d_in[3];
  const float* ln1s   = (const float*)d_in[4];
  const float* ln1b   = (const float*)d_in[5];
  const float* wq     = (const float*)d_in[6];
  const float* bq     = (const float*)d_in[7];
  const float* wk     = (const float*)d_in[8];
  const float* bk     = (const float*)d_in[9];
  const float* wv     = (const float*)d_in[10];
  const float* bv     = (const float*)d_in[11];
  const float* wo     = (const float*)d_in[12];
  const float* bo     = (const float*)d_in[13];
  const float* ln2s   = (const float*)d_in[14];
  const float* ln2b   = (const float*)d_in[15];
  const float* w1     = (const float*)d_in[16];
  const float* b1     = (const float*)d_in[17];
  const float* w2     = (const float*)d_in[18];
  const float* b2     = (const float*)d_in[19];

  float* out = (float*)d_out;
  float* out_mem  = out;                                       // [B,16,E]
  float* out_ssum = out + (size_t)B * M * E;                   // [B,E]
  float* out_x    = out + (size_t)B * M * E + (size_t)B * E;   // [B,E]

  char* ws = (char*)d_ws;
  size_t off = 0;
  auto carve = [&](size_t bytes) { char* p = ws + off; off += (bytes + 255) & ~(size_t)255; return p; };
  __bf16* wqT  = (__bf16*)carve((size_t)E * E * 2);
  __bf16* wkT  = (__bf16*)carve((size_t)E * E * 2);
  __bf16* wvT  = (__bf16*)carve((size_t)E * E * 2);
  __bf16* woT  = (__bf16*)carve((size_t)E * E * 2);
  __bf16* w1T  = (__bf16*)carve((size_t)E * F * 2);
  __bf16* w2T  = (__bf16*)carve((size_t)F * E * 2);
  __bf16* y1   = (__bf16*)carve((size_t)B * E * 2);
  __bf16* qb   = (__bf16*)carve((size_t)B * E * 2);
  __bf16* ob   = (__bf16*)carve((size_t)B * E * 2);
  __bf16* y2   = (__bf16*)carve((size_t)B * E * 2);
  __bf16* hb   = (__bf16*)carve((size_t)B * F * 2);
  float*  xmid = (float*)carve((size_t)B * E * 4);
  (void)ws_size; (void)in_sizes; (void)n_in; (void)out_size;

  // 1) weight prep (bf16, transposed to [N,K])
  int tEE = E * E, tEF = E * F;
  wtrans_kernel<<<(tEE + 255) / 256, 256, 0, stream>>>(wq, wqT, E, E);
  wtrans_kernel<<<(tEE + 255) / 256, 256, 0, stream>>>(wk, wkT, E, E);
  wtrans_kernel<<<(tEE + 255) / 256, 256, 0, stream>>>(wv, wvT, E, E);
  wtrans_kernel<<<(tEE + 255) / 256, 256, 0, stream>>>(wo, woT, E, E);
  wtrans_kernel<<<(tEF + 255) / 256, 256, 0, stream>>>(w1, w1T, E, F);
  wtrans_kernel<<<(tEF + 255) / 256, 256, 0, stream>>>(w2, w2T, F, E);

  // 2) x_in + mem shift + LN1
  pre_kernel<<<B, 256, 0, stream>>>(mem, x, delta, ln1s, ln1b, out_mem, y1);

  // 3) Q = (y1*wq + bq)/8  -> bf16
  gemm_bf16_kernel<<<dim3(B / 16, 1), 256, 0, stream>>>(
      y1, wqT, bq, E, E, 0, 0.125f, nullptr, 0, nullptr, qb, nullptr, nullptr);

  // 4) fused K/V (WMMA in LDS) + attention -> o (bf16)
  attn_kernel<<<B, 256, 0, stream>>>(out_mem, wkT, bk, wvT, bv, qb, ob);

  // 5) x_mid = o*wo + bo + x_in   (x_in = row 15 of out_mem)
  gemm_bf16_kernel<<<dim3(B / 16, 1), 256, 0, stream>>>(
      ob, woT, bo, E, E, 1, 1.f, out_mem + 15 * E, (long)(M * E), xmid, nullptr, nullptr, nullptr);

  // 6) LN2
  ln_kernel<<<B, 256, 0, stream>>>(xmid, ln2s, ln2b, y2);

  // 7) h = gelu(y2*w1 + b1)
  gemm_bf16_kernel<<<dim3(B / 16, F / 512), 256, 0, stream>>>(
      y2, w1T, b1, E, F, 2, 1.f, nullptr, 0, nullptr, hb, nullptr, nullptr);

  // 8) x_out = x_mid + h*w2 + b2 ; new_ssum = 0.9*ssum + 0.1*x_out
  gemm_bf16_kernel<<<dim3(B / 16, 1), 256, 0, stream>>>(
      hb, w2T, b2, F, E, 3, 1.f, xmid, (long)E, out_x, nullptr, ssum, out_ssum);
}
